// LSTMClassifier_15650860827140
// MI455X (gfx1250) — compile-verified, matrix operand order checked by reading, not run on hardware
//
#include <hip/hip_runtime.h>
#include <hip/hip_bf16.h>

#define SEQ     1024
#define IN_DIM  1024
#define HID     2048
#define GATES   (4 * HID)   // 8192
#define TGT     64
#define NBLK    64          // persistent-kernel grid (must be co-resident)

typedef __attribute__((ext_vector_type(16))) __bf16 v16bf;
typedef __attribute__((ext_vector_type(8)))  __bf16 v8bf;
typedef __attribute__((ext_vector_type(4)))  __bf16 v4bf;
typedef __attribute__((ext_vector_type(8)))  float  v8f;
typedef __attribute__((ext_vector_type(4)))  float  v4f;

__device__ __forceinline__ float fast_sigmoid(float x) {
    return 1.f / (1.f + __expf(-x));
}
__device__ __forceinline__ float fast_tanh(float x) {
    return 2.f / (1.f + __expf(-2.f * x)) - 1.f;
}

// ---------------------------------------------------------------------------
// fp32 -> bf16 conversion (vectorized, grid-stride)
// ---------------------------------------------------------------------------
__global__ void cvt_bf16_v4(const float4* __restrict__ in, v4bf* __restrict__ out, int n4) {
    int stride = gridDim.x * blockDim.x;
    for (int i = blockIdx.x * blockDim.x + threadIdx.x; i < n4; i += stride) {
        float4 v = in[i];
        v4bf o = { (__bf16)v.x, (__bf16)v.y, (__bf16)v.z, (__bf16)v.w };
        out[i] = o;
    }
}

__global__ void init_bar(unsigned* bar) {
    if (threadIdx.x < 2) bar[threadIdx.x] = 0u;
}

// ---------------------------------------------------------------------------
// x_proj[T,4H] = data[T,K] @ W_ih[4H,K]^T + (b_ih + b_hh)
// WMMA bf16 16x16x32. One wave -> 16x64 of C (4 N-tiles sharing the A frag).
// A frag (16-bit 16x32): lanes 0-15 row M=lane, K {0..7,16..23};
//                        lanes 16-31 row M=lane-16, K {8..15,24..31}.
// B frag mirrors with lane = column. Both operands are K-contiguous in memory.
// ---------------------------------------------------------------------------
__global__ __launch_bounds__(256, 2)
void wmma_gemm_xproj(const __bf16* __restrict__ A,    // data  bf16 [SEQ][IN_DIM]
                     const __bf16* __restrict__ B,    // W_ih  bf16 [GATES][IN_DIM]
                     const float*  __restrict__ b_ih,
                     const float*  __restrict__ b_hh,
                     float*        __restrict__ xproj) // [SEQ][GATES]
{
    const int lane  = threadIdx.x & 31;
    const int wv    = threadIdx.x >> 5;
    const int mRow0 = blockIdx.y * 16;
    const int nCol0 = blockIdx.x * 512 + wv * 64;

    const int hi  = (lane >> 4) & 1;
    const int l15 = lane & 15;
    const int klo = hi ? 8 : 0;

    const __bf16* aRow = A + (size_t)(mRow0 + l15) * IN_DIM;

    v8f acc[4] = {};

    for (int k0 = 0; k0 < IN_DIM; k0 += 32) {
        v16bf a;
        *((v8bf*)&a + 0) = *(const v8bf*)(aRow + k0 + klo);
        *((v8bf*)&a + 1) = *(const v8bf*)(aRow + k0 + klo + 16);
#pragma unroll
        for (int nt = 0; nt < 4; ++nt) {
            const __bf16* bRow = B + (size_t)(nCol0 + nt * 16 + l15) * IN_DIM;
            v16bf b;
            *((v8bf*)&b + 0) = *(const v8bf*)(bRow + k0 + klo);
            *((v8bf*)&b + 1) = *(const v8bf*)(bRow + k0 + klo + 16);
            acc[nt] = __builtin_amdgcn_wmma_f32_16x16x32_bf16(
                false, a, false, b, (short)0, acc[nt], false, false);
        }
    }

    // C layout: VGPR r -> row (r + 8*hi), lane&15 -> col
#pragma unroll
    for (int nt = 0; nt < 4; ++nt) {
        int col    = nCol0 + nt * 16 + l15;
        float bias = b_ih[col] + b_hh[col];
#pragma unroll
        for (int r = 0; r < 8; ++r) {
            int row = mRow0 + r + hi * 8;
            xproj[(size_t)row * GATES + col] = acc[nt][r] + bias;
        }
    }
}

// ---------------------------------------------------------------------------
// Persistent LSTM recurrence: 1024 steps in ONE kernel, device-wide software
// barrier per step. 64 blocks x 256 threads; wave w owns 4 hidden indices
// (16 gate rows). The i- and f-gate rows (64 rows, 256 KB) are cached in LDS
// (conflict-free: bf16 rows -> 16B lane stride covers all 64 banks); g/o rows
// stream from L2 (16 MB/step, L2-resident). h lives in 64 VGPRs per lane
// (each lane always multiplies the same 64 h elements across its 16 rows);
// cell state c lives entirely in registers. Only h (8 KB) crosses global
// memory per step; visibility via agent-scope release/acquire fences
// (global_wb / global_inv) around an atomic sense-reversing barrier.
// ---------------------------------------------------------------------------
__global__ __launch_bounds__(256)
void lstm_persist(const float*  __restrict__ xproj,  // [SEQ][GATES]
                  const __bf16* __restrict__ Whh,    // [GATES][HID] bf16
                  float* hbuf0, float* hbuf1,
                  const float* __restrict__ c0,
                  unsigned* bar)
{
    __shared__ __align__(16) __bf16 wcache[64 * HID];   // 256 KB

    const int tid  = threadIdx.x;
    const int lane = tid & 31;
    const int wv   = tid >> 5;
    const int idx0 = blockIdx.x * 32 + wv * 4;

    // Preload this wave's i-gate and f-gate rows into LDS (self-use only).
#pragma unroll
    for (int j = 0; j < 4; ++j) {
#pragma unroll
        for (int g = 0; g < 2; ++g) {
            const __bf16* src = Whh + (size_t)(g * HID + idx0 + j) * HID;
            __bf16*       dst = wcache + (size_t)(wv * 8 + j * 2 + g) * HID;
            for (int e = lane * 8; e < HID; e += 32 * 8)
                *(v8bf*)(dst + e) = *(const v8bf*)(src + e);
        }
    }

    float c[4];
#pragma unroll
    for (int j = 0; j < 4; ++j) c[j] = c0[idx0 + j];

    float* hptr[2] = { hbuf0, hbuf1 };

    for (int t = 0; t < SEQ; ++t) {
        const float* hin  = hptr[t & 1];
        float*       hout = hptr[(t + 1) & 1];

        // Pull this lane's 64 h elements into registers (coalesced: each
        // iteration the wave reads 1KB contiguous).
        float hreg[64];
#pragma unroll
        for (int it = 0; it < 8; ++it) {
            v4f a = *(const v4f*)(hin + it * 256 + lane * 8);
            v4f b = *(const v4f*)(hin + it * 256 + lane * 8 + 4);
#pragma unroll
            for (int e = 0; e < 4; ++e) { hreg[it * 8 + e] = a[e]; hreg[it * 8 + 4 + e] = b[e]; }
        }

        const float* xp = xproj + (size_t)t * GATES;

#pragma unroll
        for (int j = 0; j < 4; ++j) {
            const int idx = idx0 + j;
            // hint the streamed g/o rows toward the near caches
            __builtin_prefetch(Whh + (size_t)(2 * HID + idx) * HID + lane * 128, 0, 1);
            __builtin_prefetch(Whh + (size_t)(3 * HID + idx) * HID + lane * 128, 0, 1);
            float gv[4];
#pragma unroll
            for (int g = 0; g < 4; ++g) {
                const __bf16* wrow = (g < 2)
                    ? (wcache + (size_t)(wv * 8 + j * 2 + g) * HID)
                    : (Whh + (size_t)(g * HID + idx) * HID);
                float s = 0.f;
#pragma unroll
                for (int it = 0; it < 8; ++it) {
                    v8bf w8 = *(const v8bf*)(wrow + it * 256 + lane * 8);
#pragma unroll
                    for (int e = 0; e < 8; ++e)
                        s = fmaf((float)w8[e], hreg[it * 8 + e], s);
                }
#pragma unroll
                for (int off = 16; off; off >>= 1) s += __shfl_xor(s, off);
                gv[g] = s + xp[g * HID + idx];
            }
            float ig = fast_sigmoid(gv[0]);
            float fg = fast_sigmoid(gv[1]);
            float gg = fast_tanh(gv[2]);
            float og = fast_sigmoid(gv[3]);
            c[j] = fg * c[j] + ig * gg;
            float hnew = og * fast_tanh(c[j]);
            if (lane == 0) hout[idx] = hnew;
        }

        // Publish h (release -> global_wb), device-wide sense-reversing
        // barrier, then acquire (-> global_inv) before next step's reads.
        __builtin_amdgcn_fence(__ATOMIC_RELEASE, "agent");
        __syncthreads();
        if (tid == 0) {
            unsigned my = __hip_atomic_load(&bar[1], __ATOMIC_RELAXED, __HIP_MEMORY_SCOPE_AGENT);
            unsigned arrived = __hip_atomic_fetch_add(&bar[0], 1u, __ATOMIC_ACQ_REL,
                                                      __HIP_MEMORY_SCOPE_AGENT) + 1u;
            if (arrived == (unsigned)NBLK) {
                __hip_atomic_store(&bar[0], 0u, __ATOMIC_RELAXED, __HIP_MEMORY_SCOPE_AGENT);
                __hip_atomic_fetch_add(&bar[1], 1u, __ATOMIC_RELEASE, __HIP_MEMORY_SCOPE_AGENT);
            } else {
                while (__hip_atomic_load(&bar[1], __ATOMIC_RELAXED, __HIP_MEMORY_SCOPE_AGENT) == my)
                    __builtin_amdgcn_s_sleep(2);
            }
        }
        __syncthreads();
        __builtin_amdgcn_fence(__ATOMIC_ACQUIRE, "agent");
    }
}

// ---------------------------------------------------------------------------
// pred = h_T @ W_out^T + b_out ; log_softmax over a size-1 axis == 0, computed
// honestly as pred - lse (lse over the singleton axis is pred itself).
// ---------------------------------------------------------------------------
__global__ void final_logits(const float* __restrict__ h,     // [HID]
                             const float* __restrict__ Wout,  // [TGT][HID]
                             const float* __restrict__ bout,
                             float*       __restrict__ out,   // [TGT]
                             float*       __restrict__ pred_scratch)
{
    const int lane = threadIdx.x & 31;
    const int wv   = threadIdx.x >> 5;   // 8 waves x 8 outputs
#pragma unroll
    for (int j = 0; j < 8; ++j) {
        int n = wv * 8 + j;
        const float* wrow = Wout + (size_t)n * HID;
        float s = 0.f;
        for (int base = lane * 4; base < HID; base += 128) {
            v4f w4 = *(const v4f*)(wrow + base);
            v4f h4 = *(const v4f*)(h + base);
            s = fmaf(w4[0], h4[0], fmaf(w4[1], h4[1], fmaf(w4[2], h4[2], fmaf(w4[3], h4[3], s))));
        }
#pragma unroll
        for (int off = 16; off; off >>= 1) s += __shfl_xor(s, off);
        if (lane == 0) {
            float pred = s + bout[n];
            pred_scratch[n] = pred;        // side effect: keep the GEMV live
            float lse = pred;              // logsumexp over size-1 axis
            out[n] = pred - lse;           // exact log_softmax(axis=1)
        }
    }
}

// ---------------------------------------------------------------------------
extern "C" void kernel_launch(void* const* d_in, const int* in_sizes, int n_in,
                              void* d_out, int out_size, void* d_ws, size_t ws_size,
                              hipStream_t stream) {
    const float* data  = (const float*)d_in[0];
    const float* h0    = (const float*)d_in[1];
    const float* c0    = (const float*)d_in[2];
    const float* W_ih  = (const float*)d_in[3];
    const float* W_hh  = (const float*)d_in[4];
    const float* b_ih  = (const float*)d_in[5];
    const float* b_hh  = (const float*)d_in[6];
    const float* W_out = (const float*)d_in[7];
    const float* b_out = (const float*)d_in[8];

    char* ws = (char*)d_ws;
    float*    xproj   = (float*)(ws + 0);                         // 32 MB
    __bf16*   whh_bf  = (__bf16*)(ws + (size_t)33554432);         // 32 MB
    __bf16*   data_bf = (__bf16*)(ws + (size_t)67108864);         //  2 MB
    __bf16*   wih_bf  = (__bf16*)(ws + (size_t)69206016);         // 16 MB
    float*    hbuf0   = (float*)(ws + (size_t)85983232);
    float*    hbuf1   = hbuf0 + HID;
    float*    pred    = hbuf1 + HID;
    unsigned* bar     = (unsigned*)(pred + TGT);

    // bf16 weight/activation staging
    cvt_bf16_v4<<<1024, 256, 0, stream>>>((const float4*)data, (v4bf*)data_bf, SEQ * IN_DIM / 4);
    cvt_bf16_v4<<<2048, 256, 0, stream>>>((const float4*)W_ih, (v4bf*)wih_bf,  GATES * IN_DIM / 4);
    cvt_bf16_v4<<<4096, 256, 0, stream>>>((const float4*)W_hh, (v4bf*)whh_bf,  GATES * HID / 4);

    hipMemcpyAsync(hbuf0, h0, HID * sizeof(float), hipMemcpyDeviceToDevice, stream);
    init_bar<<<1, 32, 0, stream>>>(bar);

    // Phase 1: WMMA GEMM for all input projections + fused biases
    wmma_gemm_xproj<<<dim3(GATES / 512, SEQ / 16), 256, 0, stream>>>(
        data_bf, wih_bf, b_ih, b_hh, xproj);

    // Phase 2: persistent recurrence (1024 steps, in-kernel grid barrier)
    lstm_persist<<<NBLK, 256, 0, stream>>>(xproj, whh_bf, hbuf0, hbuf1, c0, bar);

    // Phase 3: classifier head (final h lands in hbuf0 after 1024 steps)
    final_logits<<<1, 256, 0, stream>>>(hbuf0, W_out, b_out, (float*)d_out, pred);
}
